// Agent_2894807958058
// MI455X (gfx1250) — matrix operand order repaired and checked by reading, not compiled, not run on hardware
//
#include <hip/hip_runtime.h>
#include <hip/hip_bf16.h>
#include <stdint.h>

// ---------------------------------------------------------------------------
// CDNA5 (gfx1250): convs as implicit GEMM, FC, fused LSTM gate GEMM and the
// actor+critic head all run on v_wmma_f32_16x16x32_f16 (f16 in, f32 acc).
// conv2/conv3 stage their A-tiles via global_load_async_to_lds_b128 with
// double buffering (ASYNCcnt-tracked). Weights pre-packed to fragment order.
// ---------------------------------------------------------------------------

typedef __attribute__((ext_vector_type(16))) _Float16 v16h;
typedef __attribute__((ext_vector_type(8)))  _Float16 v8h;
typedef __attribute__((ext_vector_type(8)))  float    v8f;

static __device__ __forceinline__ v16h join16(v8h lo, v8h hi) {
    return __builtin_shufflevector(lo, hi, 0,1,2,3,4,5,6,7,8,9,10,11,12,13,14,15);
}
static __device__ __forceinline__ v8f wmma_f16(v16h a, v16h b, v8f c) {
    // (neg_a, A, neg_b, B, c_mod, C, reuse_a, reuse_b)
    return __builtin_amdgcn_wmma_f32_16x16x32_f16(false, a, false, b, (short)0, c, false, false);
}
// async copy 16B global -> LDS (wave-relative LDS byte offset in low 32 bits)
static __device__ __forceinline__ void async_g2l_b128(uint32_t lds_off, const void* gaddr) {
    asm volatile("global_load_async_to_lds_b128 %0, %1, off"
                 :: "v"(lds_off), "v"((uint64_t)(uintptr_t)gaddr) : "memory");
}

// ------------------------- problem constants -------------------------------
#define TT   128
#define BB   64
#define BT   8192      // TT*BB
#define HID  512
#define NAct 15

// ------------------------- weight packing ----------------------------------
// dst[(kstep*ntiles + ntile)*512 + lane*16 + i]
//   lane 0..15  -> N = ntile*16 + lane,      K = kstep*32 + (i<8 ? i : i+8)
//   lane 16..31 -> N = ntile*16 + lane-16,   K = kstep*32 + 8 + (i<8 ? i : i+8)
__global__ __launch_bounds__(256) void pack_kernel(const float* __restrict__ s0,
                                                   const float* __restrict__ s1,
                                                   _Float16* __restrict__ dst,
                                                   int ksteps, int ntiles, int mode) {
    int idx = blockIdx.x * 256 + threadIdx.x;
    int total = ksteps * ntiles * 512;
    if (idx >= total) return;
    int per = ntiles * 512;
    int ks  = idx / per;
    int rem = idx % per;
    int nt  = rem / 512;
    int q   = rem % 512;
    int lanei = q / 16;
    int i     = q % 16;
    int k = ks * 32 + ((lanei < 16) ? 0 : 8) + ((i < 8) ? i : i + 8);
    int n = nt * 16 + (lanei & 15);
    float v = 0.0f;
    switch (mode) {
    case 0: { // conv1 weights [32,3,8,8] OIHW ; im2col k = dy*24 + dx*3 + c
        int dy = k / 24, r2 = k % 24, dx = r2 / 3, c = r2 % 3;
        v = s0[((n * 3 + c) * 8 + dy) * 8 + dx];
    } break;
    case 1: { // conv2 weights [64,32,4,4] ; k = (dy*4+dx)*32 + c
        int dy = k / 128, r2 = k % 128, dx = r2 / 32, c = r2 % 32;
        v = s0[((n * 32 + c) * 4 + dy) * 4 + dx];
    } break;
    case 2: { // conv3 weights [64,64,3,3] ; k = (dy*3+dx)*64 + c
        int dy = k / 192, r2 = k % 192, dx = r2 / 64, c = r2 % 64;
        v = s0[((n * 64 + c) * 3 + dy) * 3 + dx];
    } break;
    case 3: { // fc_w [576,512]; our a3 is NHWC-flat (y,x,c); ref flatten is (c,y,x)
        int sidx = k / 64, c = k % 64;
        int kref = c * 9 + sidx;
        v = s0[kref * HID + n];
    } break;
    case 4: { // LSTM: B[k][n] = k<512 ? w_ih[n][k] : w_hh[n][k-512]
        v = (k < HID) ? s0[n * HID + k] : s1[n * HID + (k - HID)];
    } break;
    case 5: { // head: cols 0..14 actor_w [512,15], col 15 critic_w [512,1]
        v = (n < NAct) ? s0[k * NAct + n] : s1[k];
    } break;
    }
    dst[idx] = (_Float16)v;
}

// ------------------------- conv1 (f32 input, /255) -------------------------
// im2col GEMM: M = BT*144, K = 192 (6 ksteps), N = 32 (2 ntiles per wave)
__global__ __launch_bounds__(128) void conv1_kernel(const float* __restrict__ x,
                                                    const _Float16* __restrict__ Bp,
                                                    const float* __restrict__ bias,
                                                    _Float16* __restrict__ out) {
    __shared__ _Float16 sA[4][16 * 40];
    int lane = threadIdx.x & 31, wv = threadIdx.x >> 5;
    int gw = blockIdx.x * 4 + wv;
    int mbase = gw * 16;
    // hoist per-row input base offsets out of the K loop
    int ib[16];
#pragma unroll
    for (int i = 0; i < 16; ++i) {
        int m = mbase + i;
        int nimg = m / 144, rem = m % 144;
        int oy = rem / 12, ox = rem % 12;
        ib[i] = ((nimg * 52 + oy * 4) * 52 + ox * 4) * 3;
    }
    v8f acc[2] = {};
    for (int ks = 0; ks < 6; ++ks) {
        // lane stages im2col column k = ks*32+lane for all 16 rows
        int k = ks * 32 + lane;
        int dy = k / 24, t2 = k % 24, dx = t2 / 3, c = t2 % 3;
        int off = (dy * 52 + dx) * 3 + c;
#pragma unroll
        for (int i = 0; i < 16; ++i)
            sA[wv][i * 40 + lane] = (_Float16)(x[ib[i] + off] * (1.0f / 255.0f));
        __syncthreads();
        int r = lane & 15;
        int k0 = (lane < 16) ? 0 : 8;
        v8h lo = *(const v8h*)&sA[wv][r * 40 + k0];
        v8h hi = *(const v8h*)&sA[wv][r * 40 + k0 + 16];
        v16h a = join16(lo, hi);
        for (int nt = 0; nt < 2; ++nt) {
            v16h b = *(const v16h*)&Bp[(ks * 2 + nt) * 512 + lane * 16];
            acc[nt] = wmma_f16(a, b, acc[nt]);
        }
        __syncthreads();
    }
    int rowoff = (lane < 16) ? 0 : 8;
    for (int nt = 0; nt < 2; ++nt)
        for (int v = 0; v < 8; ++v) {
            int m = mbase + rowoff + v;
            int n = nt * 16 + (lane & 15);
            float val = acc[nt][v] + bias[n];
            out[m * 32 + n] = (_Float16)fmaxf(val, 0.0f);
        }
}

// ------------------------- conv2 / conv3 (f16 input) -----------------------
// NHWC implicit GEMM, OC=64 (4 ntiles per wave). Each (row,kstep) A-slice is
// 64B contiguous in global (C multiple of 32) -> staged with
// global_load_async_to_lds_b128, double-buffered on ASYNCcnt.
__global__ __launch_bounds__(128) void convf16_kernel(const _Float16* __restrict__ in,
                                                      const _Float16* __restrict__ Bp,
                                                      const float* __restrict__ bias,
                                                      _Float16* __restrict__ out,
                                                      int H, int W, int C,
                                                      int OH, int OW, int stride,
                                                      int KW_C, int ksteps) {
    __shared__ _Float16 sA[4][2][16 * 40];   // per-wave double buffer
    int lane = threadIdx.x & 31, wv = threadIdx.x >> 5;
    int gw = blockIdx.x * 4 + wv;
    int mbase = gw * 16;
    int OHW = OH * OW;

    // lane services transfers idx = lane and lane+32:  r = idx>>2, seg = idx&3
    int seg = lane & 3;
    int r0 = lane >> 2;        // rows 0..7
    int r1 = 8 + (lane >> 2);  // rows 8..15
    const _Float16* rb[2];
#pragma unroll
    for (int j = 0; j < 2; ++j) {
        int m = mbase + (j ? r1 : r0);
        int nimg = m / OHW, rem = m % OHW;
        int oy = rem / OW, ox = rem % OW;
        rb[j] = in + ((nimg * H + oy * stride) * W + ox * stride) * C;
    }
    uint32_t ldsbase = (uint32_t)(uintptr_t)&sA[wv][0][0];   // wave-relative LDS offset

    auto issue = [&](int ks, int buf) {
        int k = ks * 32;
        int dy = k / KW_C, r2 = k % KW_C, dx = r2 / C, c0 = r2 % C;
        int off = (dy * W + dx) * C + c0;                    // halves
        uint32_t lb = ldsbase + (uint32_t)buf * 1280u + (uint32_t)seg * 16u;
        async_g2l_b128(lb + (uint32_t)r0 * 80u, rb[0] + off + seg * 8);
        async_g2l_b128(lb + (uint32_t)r1 * 80u, rb[1] + off + seg * 8);
    };

    v8f acc[4] = {};
    issue(0, 0);
    int r = lane & 15;
    int k0 = (lane < 16) ? 0 : 8;
    for (int ks = 0; ks < ksteps; ++ks) {
        if (ks + 1 < ksteps) {
            issue(ks + 1, (ks + 1) & 1);
            asm volatile("s_wait_asynccnt 2" ::: "memory");  // current buffer landed
        } else {
            asm volatile("s_wait_asynccnt 0" ::: "memory");
        }
        const _Float16* sb = &sA[wv][ks & 1][0];
        v8h lo = *(const v8h*)&sb[r * 40 + k0];
        v8h hi = *(const v8h*)&sb[r * 40 + k0 + 16];
        v16h a = join16(lo, hi);
        for (int nt = 0; nt < 4; ++nt) {
            v16h b = *(const v16h*)&Bp[(ks * 4 + nt) * 512 + lane * 16];
            acc[nt] = wmma_f16(a, b, acc[nt]);
        }
    }
    int rowoff = (lane < 16) ? 0 : 8;
    for (int nt = 0; nt < 4; ++nt)
        for (int v = 0; v < 8; ++v) {
            int m = mbase + rowoff + v;
            int n = nt * 16 + (lane & 15);
            float val = acc[nt][v] + bias[n];
            out[m * 64 + n] = (_Float16)fmaxf(val, 0.0f);
        }
}

// ------------------------- FC: [BT,576] @ [576,512] + relu -----------------
__global__ __launch_bounds__(128) void fc_gemm_kernel(const _Float16* __restrict__ A,
                                                      const _Float16* __restrict__ Bp,
                                                      const float* __restrict__ bias,
                                                      _Float16* __restrict__ out) {
    int lane = threadIdx.x & 31, wv = threadIdx.x >> 5;
    int w = blockIdx.x * 4 + wv;
    int mtile = w >> 3, ng = w & 7;
    int mbase = mtile * 16, nt0 = ng * 4;
    int r = lane & 15;
    int m = mbase + r;
    int k0 = (lane < 16) ? 0 : 8;
    v8f acc[4] = {};
    for (int ks = 0; ks < 18; ++ks) {
        const _Float16* p = A + m * 576 + ks * 32;
        __builtin_prefetch(p + 32, 0, 3);
        v8h lo = *(const v8h*)(p + k0);
        v8h hi = *(const v8h*)(p + k0 + 16);
        v16h a = join16(lo, hi);
        for (int nt = 0; nt < 4; ++nt) {
            v16h b = *(const v16h*)&Bp[(ks * 32 + nt0 + nt) * 512 + lane * 16];
            acc[nt] = wmma_f16(a, b, acc[nt]);
        }
    }
    int rowoff = (lane < 16) ? 0 : 8;
    for (int nt = 0; nt < 4; ++nt)
        for (int v = 0; v < 8; ++v) {
            int mm = mbase + rowoff + v;
            int n = (nt0 + nt) * 16 + (lane & 15);
            float val = acc[nt][v] + bias[n];
            out[mm * HID + n] = (_Float16)fmaxf(val, 0.0f);
        }
}

// ------------------- LSTM gate GEMM: [64,1024] @ [1024,2048] ---------------
// A row b = concat(x_t[b], (1-done_t[b])*h[b]); 32 ksteps; 128 ntiles.
__global__ __launch_bounds__(128) void lstm_gemm_kernel(const _Float16* __restrict__ xt,
                                                        const _Float16* __restrict__ hf,
                                                        const float* __restrict__ dvec,
                                                        const _Float16* __restrict__ Bp,
                                                        float* __restrict__ gates) {
    int lane = threadIdx.x & 31, wv = threadIdx.x >> 5;
    int w = blockIdx.x * 4 + wv;
    int mtile = w >> 5, ng = w & 31;
    int mbase = mtile * 16, nt0 = ng * 4;
    int r = lane & 15;
    int b = mbase + r;
    int k0 = (lane < 16) ? 0 : 8;
    _Float16 mh = (_Float16)(1.0f - dvec[b]);
    v8f acc[4] = {};
    for (int ks = 0; ks < 32; ++ks) {
        v8h lo, hi;
        if (ks < 16) {
            const _Float16* p = xt + b * HID + ks * 32;
            lo = *(const v8h*)(p + k0);
            hi = *(const v8h*)(p + k0 + 16);
        } else {
            const _Float16* p = hf + b * HID + (ks - 16) * 32;
            lo = *(const v8h*)(p + k0) * mh;   // fold done-mask into h operand
            hi = *(const v8h*)(p + k0 + 16) * mh;
        }
        v16h a = join16(lo, hi);
        const _Float16* bp = &Bp[(ks * 128 + nt0) * 512 + lane * 16];
        __builtin_prefetch(bp + 128 * 512, 0, 3);
        for (int nt = 0; nt < 4; ++nt) {
            v16h bb = *(const v16h*)(bp + nt * 512);
            acc[nt] = wmma_f16(a, bb, acc[nt]);
        }
    }
    int rowoff = (lane < 16) ? 0 : 8;
    for (int nt = 0; nt < 4; ++nt)
        for (int v = 0; v < 8; ++v) {
            int bb2 = mbase + rowoff + v;
            int n = (nt0 + nt) * 16 + (lane & 15);
            gates[bb2 * 2048 + n] = acc[nt][v];
        }
}

// ------------------------- LSTM elementwise update -------------------------
__global__ __launch_bounds__(256) void lstm_update_kernel(const float* __restrict__ gates,
                                                          const float* __restrict__ bl,
                                                          const float* __restrict__ dvec,
                                                          float* __restrict__ hst,
                                                          float* __restrict__ cst,
                                                          _Float16* __restrict__ hf,
                                                          _Float16* __restrict__ ft) {
    int idx = blockIdx.x * 256 + threadIdx.x;   // 64*512
    int b = idx >> 9, j = idx & 511;
    float m = 1.0f - dvec[b];
    const float* g = gates + b * 2048;
    float ig = g[j]        + bl[j];
    float fg = g[512 + j]  + bl[512 + j];
    float gg = g[1024 + j] + bl[1024 + j];
    float og = g[1536 + j] + bl[1536 + j];
    float si = 1.0f / (1.0f + expf(-ig));
    float sf = 1.0f / (1.0f + expf(-fg));
    float so = 1.0f / (1.0f + expf(-og));
    float c = sf * (m * cst[idx]) + si * tanhf(gg);
    float h = so * tanhf(c);
    cst[idx] = c;
    hst[idx] = h;
    hf[idx] = (_Float16)h;
    ft[idx] = (_Float16)h;
}

__global__ __launch_bounds__(256) void lstm_init_kernel(const float* __restrict__ h0,
                                                        const float* __restrict__ c0,
                                                        float* __restrict__ hst,
                                                        float* __restrict__ cst,
                                                        _Float16* __restrict__ hf) {
    int idx = blockIdx.x * 256 + threadIdx.x;
    hst[idx] = h0[idx];
    cst[idx] = c0[idx];
    hf[idx] = (_Float16)h0[idx];
}

// --------------- head GEMM: feats [BT,512] @ [512,16] (actor|critic) -------
__global__ __launch_bounds__(128) void head_gemm_kernel(const _Float16* __restrict__ feats,
                                                        const _Float16* __restrict__ Bp,
                                                        float* __restrict__ out16) {
    int lane = threadIdx.x & 31, wv = threadIdx.x >> 5;
    int w = blockIdx.x * 4 + wv;
    int mbase = w * 16;
    int r = lane & 15;
    int m = mbase + r;
    int k0 = (lane < 16) ? 0 : 8;
    v8f acc = {};
    for (int ks = 0; ks < 16; ++ks) {
        const _Float16* p = feats + m * HID + ks * 32;
        __builtin_prefetch(p + 32, 0, 3);
        v8h lo = *(const v8h*)(p + k0);
        v8h hi = *(const v8h*)(p + k0 + 16);
        v16h a = join16(lo, hi);
        v16h b = *(const v16h*)&Bp[ks * 512 + lane * 16];
        acc = wmma_f16(a, b, acc);
    }
    int rowoff = (lane < 16) ? 0 : 8;
    for (int v = 0; v < 8; ++v)
        out16[(mbase + rowoff + v) * 16 + (lane & 15)] = acc[v];
}

// ------------------- log-softmax / gather / entropy / value ----------------
__global__ __launch_bounds__(256) void post_kernel(const float* __restrict__ out16,
                                                   const float* __restrict__ actor_b,
                                                   const float* __restrict__ critic_b,
                                                   const int* __restrict__ action,
                                                   float* __restrict__ out) {
    int r = blockIdx.x * 256 + threadIdx.x;
    if (r >= BT) return;
    float l[NAct];
    float mx = -1e30f;
    for (int j = 0; j < NAct; ++j) {
        l[j] = out16[r * 16 + j] + actor_b[j];
        mx = fmaxf(mx, l[j]);
    }
    float s = 0.0f;
    for (int j = 0; j < NAct; ++j) s += expf(l[j] - mx);
    float lse = mx + logf(s);
    int a = action[r];
    out[r] = l[a] - lse;
    float ent = 0.0f;
    for (int j = 0; j < NAct; ++j) {
        float lp = l[j] - lse;
        ent -= expf(lp) * lp;
    }
    out[BT + r] = ent;
    out[2 * BT + r] = out16[r * 16 + NAct] + critic_b[0];
}

__global__ __launch_bounds__(256) void final_state_kernel(const float* __restrict__ hst,
                                                          const float* __restrict__ cst,
                                                          float* __restrict__ out) {
    int idx = blockIdx.x * 256 + threadIdx.x;    // 32768
    out[3 * BT + idx] = hst[idx];
    out[3 * BT + BB * HID + idx] = cst[idx];
}

// ---------------------------------------------------------------------------
extern "C" void kernel_launch(void* const* d_in, const int* in_sizes, int n_in,
                              void* d_out, int out_size, void* d_ws, size_t ws_size,
                              hipStream_t stream) {
    const float* x      = (const float*)d_in[0];
    const float* done   = (const float*)d_in[1];
    const float* h0     = (const float*)d_in[2];
    const float* c0     = (const float*)d_in[3];
    const int*   action = (const int*)  d_in[4];
    const float* cw1 = (const float*)d_in[5],  *cb1 = (const float*)d_in[6];
    const float* cw2 = (const float*)d_in[7],  *cb2 = (const float*)d_in[8];
    const float* cw3 = (const float*)d_in[9],  *cb3 = (const float*)d_in[10];
    const float* fcw = (const float*)d_in[11], *fcb = (const float*)d_in[12];
    const float* wih = (const float*)d_in[13], *whh = (const float*)d_in[14];
    const float* blstm = (const float*)d_in[15];
    const float* aw  = (const float*)d_in[16], *abias = (const float*)d_in[17];
    const float* crw = (const float*)d_in[18], *crb = (const float*)d_in[19];
    float* out = (float*)d_out;

    // ---- workspace layout (256B aligned slices) ----
    char* base = (char*)d_ws;
    size_t off = 0;
    auto alloc = [&](size_t bytes) -> char* {
        char* p = base + off;
        off += (bytes + 255) & ~size_t(255);
        return p;
    };
    _Float16* a1     = (_Float16*)alloc((size_t)BT * 144 * 32 * 2); // conv1 out NHWC
    _Float16* a2     = (_Float16*)alloc((size_t)BT * 25 * 64 * 2);  // conv2 out
    _Float16* a3     = (_Float16*)alloc((size_t)BT * 9 * 64 * 2);   // conv3 out
    _Float16* hidden = (_Float16*)alloc((size_t)BT * HID * 2);      // FC out (LSTM x_t)
    _Float16* feats  = (_Float16*)alloc((size_t)BT * HID * 2);      // LSTM hs (head input)
    _Float16* hf16   = (_Float16*)alloc((size_t)BB * HID * 2);
    float*    hst    = (float*)alloc((size_t)BB * HID * 4);
    float*    cst    = (float*)alloc((size_t)BB * HID * 4);
    float*    gates  = (float*)alloc((size_t)BB * 2048 * 4);
    float*    out16  = (float*)alloc((size_t)BT * 16 * 4);
    _Float16* pC1    = (_Float16*)alloc(6  * 2   * 512 * 2);
    _Float16* pC2    = (_Float16*)alloc(16 * 4   * 512 * 2);
    _Float16* pC3    = (_Float16*)alloc(18 * 4   * 512 * 2);
    _Float16* pFC    = (_Float16*)alloc(18 * 32  * 512 * 2);
    _Float16* pLSTM  = (_Float16*)alloc(32 * 128 * 512 * 2);
    _Float16* pHEAD  = (_Float16*)alloc(16 * 1   * 512 * 2);

    // ---- weight packing ----
    pack_kernel<<<(6 * 2 * 512 + 255) / 256, 256, 0, stream>>>(cw1, nullptr, pC1, 6, 2, 0);
    pack_kernel<<<(16 * 4 * 512 + 255) / 256, 256, 0, stream>>>(cw2, nullptr, pC2, 16, 4, 1);
    pack_kernel<<<(18 * 4 * 512 + 255) / 256, 256, 0, stream>>>(cw3, nullptr, pC3, 18, 4, 2);
    pack_kernel<<<(18 * 32 * 512 + 255) / 256, 256, 0, stream>>>(fcw, nullptr, pFC, 18, 32, 3);
    pack_kernel<<<(32 * 128 * 512 + 255) / 256, 256, 0, stream>>>(wih, whh, pLSTM, 32, 128, 4);
    pack_kernel<<<(16 * 1 * 512 + 255) / 256, 256, 0, stream>>>(aw, crw, pHEAD, 16, 1, 5);

    // ---- CNN encoder ----
    conv1_kernel<<<18432, 128, 0, stream>>>(x, pC1, cb1, a1);                 // M=BT*144
    convf16_kernel<<<3200, 128, 0, stream>>>(a1, pC2, cb2, a2,
                                             12, 12, 32, 5, 5, 2, 4 * 32, 16); // M=BT*25
    convf16_kernel<<<1152, 128, 0, stream>>>(a2, pC3, cb3, a3,
                                             5, 5, 64, 3, 3, 1, 3 * 64, 18);   // M=BT*9
    fc_gemm_kernel<<<1024, 128, 0, stream>>>(a3, pFC, fcb, hidden);            // 4096 waves

    // ---- LSTM scan (T sequential steps) ----
    lstm_init_kernel<<<(BB * HID) / 256, 256, 0, stream>>>(h0, c0, hst, cst, hf16);
    for (int t = 0; t < TT; ++t) {
        const _Float16* xt = hidden + (size_t)t * BB * HID;
        const float* dt = done + t * BB;
        lstm_gemm_kernel<<<32, 128, 0, stream>>>(xt, hf16, dt, pLSTM, gates);
        lstm_update_kernel<<<(BB * HID) / 256, 256, 0, stream>>>(
            gates, blstm, dt, hst, cst, hf16, feats + (size_t)t * BB * HID);
    }

    // ---- heads ----
    head_gemm_kernel<<<BT / 16 / 4, 128, 0, stream>>>(feats, pHEAD, out16);
    post_kernel<<<BT / 256, 256, 0, stream>>>(out16, abias, crb, action, out);
    final_state_kernel<<<(BB * HID) / 256, 256, 0, stream>>>(hst, cst, out);

    (void)in_sizes; (void)n_in; (void)out_size; (void)ws_size;
}
